// OptimizedSelfAttention_6889127543339
// MI455X (gfx1250) — matrix operand order repaired and checked by reading, not compiled, not run on hardware
//
#include <hip/hip_runtime.h>
#include <math.h>

// ---- sizes (fixed by the reference) ----
#define Bb  4
#define Cc  256
#define C8n 32
#define Nn  16384          // H*W = 128*128
#define CN  (Cc * Nn)

typedef __attribute__((ext_vector_type(8)))  float          v8f;
typedef __attribute__((ext_vector_type(4)))  float          f4;
typedef __attribute__((ext_vector_type(16))) __bf16         v16bf;
typedef __attribute__((ext_vector_type(8)))  __bf16         v8bf;
typedef __attribute__((ext_vector_type(8)))  unsigned short v8u16;
typedef __attribute__((ext_vector_type(2)))  unsigned       u32x2;
typedef __attribute__((ext_vector_type(4)))  unsigned       u32x4;
typedef __attribute__((ext_vector_type(8)))  unsigned       u32x8;
typedef __attribute__((ext_vector_type(4)))  int            i32x4;
typedef __attribute__((ext_vector_type(8)))  int            i32x8;

// round-to-nearest (+0x8000) then pack the two high halves with one v_perm_b32
__device__ __forceinline__ unsigned pk_bf16x2(float lo, float hi) {
    unsigned a = __builtin_bit_cast(unsigned, lo) + 0x8000u;
    unsigned b = __builtin_bit_cast(unsigned, hi) + 0x8000u;
    return __builtin_amdgcn_perm(b, a, 0x07060302u);  // {b.hi16, a.hi16}
}
__device__ __forceinline__ unsigned short bf16bits(float f) {   // scalar helper
    unsigned u = __builtin_bit_cast(unsigned, f) + 0x8000u;
    return (unsigned short)(u >> 16);
}
__device__ __forceinline__ v8bf bcast8(v8u16 v) { return __builtin_bit_cast(v8bf, v); }
__device__ __forceinline__ v16bf cat16(v8bf lo, v8bf hi) {
    return __builtin_shufflevector(lo, hi, 0,1,2,3,4,5,6,7,8,9,10,11,12,13,14,15);
}
__device__ __forceinline__ v16bf cvt16(f4 a0, f4 a1, f4 a2, f4 a3) {
    u32x8 w;
    w[0] = pk_bf16x2(a0[0], a0[1]); w[1] = pk_bf16x2(a0[2], a0[3]);
    w[2] = pk_bf16x2(a1[0], a1[1]); w[3] = pk_bf16x2(a1[2], a1[3]);
    w[4] = pk_bf16x2(a2[0], a2[1]); w[5] = pk_bf16x2(a2[2], a2[3]);
    w[6] = pk_bf16x2(a3[0], a3[1]); w[7] = pk_bf16x2(a3[2], a3[3]);
    return __builtin_bit_cast(v16bf, w);
}

#define LDS_CP 264   // padded row stride (halves): 264*2B = 528B = 33*16B

// ---------------------------------------------------------------------------
// P1: kT[b][n][o] = (Wk x)[o,n] + bk[o]   via WMMA.  A = X^T (LDS), B = Wk^T.
// grid: B * N/64 = 1024 blocks, 256 threads (8 waves).
// ---------------------------------------------------------------------------
__global__ __launch_bounds__(256) void k_proj_k(
    const float* __restrict__ x, const float* __restrict__ Wk,
    const float* __restrict__ bk, float* __restrict__ kT)
{
    __shared__ __align__(16) unsigned short xs[64 * LDS_CP];
    const int tid   = threadIdx.x;
    const int blk   = blockIdx.x;
    const int b     = blk >> 8;
    const int nbase = (blk & 255) << 6;
    const float* xb = x + (size_t)b * CN;
    {   // stage x[b, :, nbase..+63] transposed -> xs[n][c] bf16, ds_store_b64 packed
        const int n = tid & 63, cq = tid >> 6;          // cq 0..3
        const float* base = xb + nbase + n;
        u32x2* xs2 = (u32x2*)xs;                        // 66 u32x2 per row
#pragma unroll
        for (int j = 0; j < 16; ++j) {
            const int q = cq + (j << 2);                // quad index 0..63, c = 4q
            const float* p = base + (size_t)(q << 2) * Nn;
            u32x2 v;
            v[0] = pk_bf16x2(p[0],          p[Nn]);
            v[1] = pk_bf16x2(p[2 * Nn],     p[3 * Nn]);
            xs2[n * 66 + q] = v;
        }
    }
    __syncthreads();

    const int w  = tid >> 5, L = tid & 31;
    const int nt = w >> 1, ot = w & 1;
    const int aoff = (L >> 4) << 3;                 // 0 or 8  (unused; A uses koff)
    const int koff = (L >> 4) << 4;                 // 0 or 16 (B K-half)
    const int arow = (nt * 16 + (L & 15)) * LDS_CP;
    const v8u16* lv = (const v8u16*)xs;
    const float* wrow = Wk + (size_t)(ot * 16 + (L & 15)) * 256;

    v8f acc = 0.f;
#pragma unroll
    for (int ks = 0; ks < 8; ++ks) {
        const int k0 = ks << 5;
        const int h0 = arow + k0 + koff;
        v16bf A = cat16(bcast8(lv[h0 >> 3]), bcast8(lv[(h0 >> 3) + 1]));
        const f4* wr = (const f4*)(wrow + k0 + koff);
        v16bf Bm = cvt16(wr[0], wr[1], wr[2], wr[3]);
        acc = __builtin_amdgcn_wmma_f32_16x16x32_bf16(false, A, false, Bm,
                                                      (short)0, acc, false, false);
    }
    const int o = ot * 16 + (L & 15);
    const float bko = bk[o];
    float* kb = kT + (size_t)b * Nn * 32;
#pragma unroll
    for (int r = 0; r < 8; ++r) {
        int m = r + ((L >> 4) << 3);
        int n = nbase + nt * 16 + m;
        kb[(size_t)n * 32 + o] = acc[r] + bko;
    }
}

// ---------------------------------------------------------------------------
// P2a: per-(b,o) softmax max & 1/sum over N.  grid: B blocks, 256 threads.
// ---------------------------------------------------------------------------
__global__ __launch_bounds__(256) void k_softmax_stats(
    const float* __restrict__ kT, float* __restrict__ rmax, float* __restrict__ rinv)
{
    __shared__ float red[256];
    __shared__ float sm[32];
    const int b = blockIdx.x, t = threadIdx.x;
    const int o = t & 31, seg = t >> 5;
    const float* kb = kT + (size_t)b * Nn * 32;

    float m = -3.4e38f;
    for (int n = seg * 2048; n < seg * 2048 + 2048; ++n)
        m = fmaxf(m, kb[(size_t)n * 32 + o]);
    red[t] = m; __syncthreads();
    if (t < 32) {
        float mm = red[t];
#pragma unroll
        for (int i = 1; i < 8; ++i) mm = fmaxf(mm, red[i * 32 + t]);
        sm[t] = mm;
    }
    __syncthreads();
    m = sm[o];
    float s = 0.f;
    for (int n = seg * 2048; n < seg * 2048 + 2048; ++n)
        s += expf(kb[(size_t)n * 32 + o] - m);
    __syncthreads();
    red[t] = s; __syncthreads();
    if (t < 32) {
        float ss = red[t];
#pragma unroll
        for (int i = 1; i < 8; ++i) ss += red[i * 32 + t];
        rmax[b * 32 + t] = sm[t];
        rinv[b * 32 + t] = 1.0f / ss;
    }
}

// ---------------------------------------------------------------------------
// P2b: wT[b][o][n] = bf16( exp(k - max) * invsum ).  grid: 1024 blocks.
// ---------------------------------------------------------------------------
__global__ __launch_bounds__(256) void k_softmax_norm(
    const float* __restrict__ kT, const float* __restrict__ rmax,
    const float* __restrict__ rinv, unsigned short* __restrict__ wT)
{
    const int gid  = blockIdx.x * 256 + threadIdx.x;
    const int base = gid * 8;                 // over B*32*N elements
    const int n0   = base & (Nn - 1);
    const int bo   = base >> 14;              // b*32 + o
    const int o    = bo & 31, b = bo >> 5;
    const float m = rmax[bo], iv = rinv[bo];
    const float* kb = kT + (size_t)b * Nn * 32;
    float e[8];
#pragma unroll
    for (int i = 0; i < 8; ++i)
        e[i] = expf(kb[(size_t)(n0 + i) * 32 + o] - m) * iv;
    u32x4 outv;
#pragma unroll
    for (int i = 0; i < 4; ++i) outv[i] = pk_bf16x2(e[2 * i], e[2 * i + 1]);
    *(u32x4*)(wT + (size_t)bo * Nn + n0) = outv;
}

// ---------------------------------------------------------------------------
// P3: tT[b][c][o] += sum_n w[o,n]*x[c,n]  (split-K over N, WMMA, f32 atomics)
// 512 wave-jobs: (b, c-tile of 16, n-slice of 2048).  grid: 64 blocks.
// ---------------------------------------------------------------------------
__global__ __launch_bounds__(256) void k_kv_part1(
    const float* __restrict__ x, const unsigned short* __restrict__ wT,
    float* __restrict__ tT)
{
    const int tid = threadIdx.x;
    const int wg  = blockIdx.x * 8 + (tid >> 5);
    const int L   = tid & 31;
    const int b   = wg >> 7;
    const int rem = wg & 127;
    const int cb     = (rem >> 3) << 4;
    const int nstart = (rem & 7) << 11;

    const float* xr = x + (size_t)b * CN + (size_t)(cb + (L & 15)) * Nn;
    const unsigned short* wp = wT + (size_t)b * 32 * Nn;
    const int aoff = (L >> 4) << 3;
    const int boff = (L >> 4) << 4;
    const int o0 = (L & 15), o1 = 16 + (L & 15);

    v8f acc0 = 0.f, acc1 = 0.f;
    for (int ks = 0; ks < 64; ++ks) {
        const int n0 = nstart + (ks << 5);
        const f4* xp = (const f4*)(xr + n0 + boff);
        v16bf A = cvt16(xp[0], xp[1], xp[2], xp[3]);
        const v8u16* b0p = (const v8u16*)(wp + (size_t)o0 * Nn + n0 + boff);
        const v8u16* b1p = (const v8u16*)(wp + (size_t)o1 * Nn + n0 + boff);
        v16bf B0 = cat16(bcast8(b0p[0]), bcast8(b0p[1]));
        v16bf B1 = cat16(bcast8(b1p[0]), bcast8(b1p[1]));
        acc0 = __builtin_amdgcn_wmma_f32_16x16x32_bf16(false, A, false, B0,
                                                       (short)0, acc0, false, false);
        acc1 = __builtin_amdgcn_wmma_f32_16x16x32_bf16(false, A, false, B1,
                                                       (short)0, acc1, false, false);
    }
    float* tb = tT + (size_t)b * 8192;
#pragma unroll
    for (int r = 0; r < 8; ++r) {
        int m = r + ((L >> 4) << 3);
        int c = cb + m;
        unsafeAtomicAdd(&tb[c * 32 + o0], acc0[r]);
        unsafeAtomicAdd(&tb[c * 32 + o1], acc1[r]);
    }
}

// ---------------------------------------------------------------------------
// P4a: kv[b][o][c] = sum_c' Wv[c,c'] * tT[b][c'][o] + bv[c].  grid: B*8.
// ---------------------------------------------------------------------------
__global__ __launch_bounds__(256) void k_kv_part2(
    const float* __restrict__ tT, const float* __restrict__ Wv,
    const float* __restrict__ bv, float* __restrict__ kv)
{
    __shared__ float st[256 * 32];
    __shared__ float sw[32 * 260];
    const int blk = blockIdx.x;
    const int b = blk >> 3, cchunk = (blk & 7) << 5;
    const int t = threadIdx.x;
    const float* tb = tT + (size_t)b * 8192;
    for (int i = t; i < 8192; i += 256) st[i] = tb[i];
    for (int i = t; i < 8192; i += 256)
        sw[(i >> 8) * 260 + (i & 255)] = Wv[(size_t)(cchunk + (i >> 8)) * 256 + (i & 255)];
    __syncthreads();
    for (int u = t; u < 1024; u += 256) {
        const int o = u >> 5, ci = u & 31;
        const int c = cchunk + ci;
        float s = bv[c];
        const float* wr = &sw[ci * 260];
        for (int cc = 0; cc < 256; ++cc)
            s += wr[cc] * st[cc * 32 + o];
        kv[(size_t)b * 8192 + o * 256 + c] = s;
    }
}

// ---------------------------------------------------------------------------
// P4b: Mt[b][c][c'] = bf16( sum_o Wq[o,c'] * kv[b][o][c] ); d[b][c] = bq·kv.
// grid: B*16 blocks (16 c's each).
// ---------------------------------------------------------------------------
__global__ __launch_bounds__(256) void k_form_M(
    const float* __restrict__ kv, const float* __restrict__ Wq,
    const float* __restrict__ bq, unsigned short* __restrict__ Mt,
    float* __restrict__ dvec)
{
    __shared__ float swq[32 * 260];
    __shared__ float kvc[32 * 16];
    const int blk = blockIdx.x;
    const int b = blk >> 4, cchunk = (blk & 15) << 4;
    const int t = threadIdx.x;
    for (int i = t; i < 8192; i += 256)
        swq[(i >> 8) * 260 + (i & 255)] = Wq[i];
    for (int i = t; i < 512; i += 256) {
        int o = i >> 4, ci = i & 15;
        kvc[o * 16 + ci] = kv[(size_t)b * 8192 + o * 256 + cchunk + ci];
    }
    __syncthreads();
    if (t < 16) {
        float s = 0.f;
        for (int o = 0; o < 32; ++o) s += bq[o] * kvc[o * 16 + t];
        dvec[b * 256 + cchunk + t] = s;
    }
    const int ci = t >> 4, cpb = (t & 15) << 4;
#pragma unroll
    for (int j = 0; j < 16; ++j) {
        const int cp = cpb + j;
        float s = 0.f;
#pragma unroll
        for (int o = 0; o < 32; ++o)
            s += swq[o * 260 + cp] * kvc[o * 16 + ci];
        Mt[(size_t)b * 65536 + (size_t)(cchunk + ci) * 256 + cp] = bf16bits(s);
    }
}

// ---------------------------------------------------------------------------
// P5: out = gamma * (X^T · M + d) + x.  Mt staged into LDS via the Tensor
// Data Mover (tensor_load_to_lds, TENSORcnt), padded to a 264-half stride.
// grid: B * N/32 = 2048 blocks, 8 waves; 32 WMMAs/wave.
// ---------------------------------------------------------------------------
__global__ __launch_bounds__(256) void k_attn_out(
    const float* __restrict__ x, const unsigned short* __restrict__ Mt,
    const float* __restrict__ dvec, const float* __restrict__ gamma,
    float* __restrict__ out)
{
    __shared__ __align__(16) unsigned short xs[32 * LDS_CP];    // 16,896 B
    __shared__ __align__(16) unsigned short sMt[256 * LDS_CP];  // 135,168 B
    const int tid   = threadIdx.x;
    const int blk   = blockIdx.x;
    const int b     = blk >> 9;                 // 512 tiles per batch
    const int nbase = (blk & 511) << 5;         // 32 n per tile
    const float* xb = x + (size_t)b * CN;
    const unsigned short* mb = Mt + (size_t)b * 65536;

    // ---- stage Mt[256][256] bf16 -> sMt (row stride 264 halves) ----
#if __has_builtin(__builtin_amdgcn_tensor_load_to_lds)
    if (tid < 32) {   // wave 0 issues one TDM descriptor for the whole tile
        const unsigned lds_addr = (unsigned)(size_t)(&sMt[0]);
        const unsigned long long ga = (unsigned long long)(size_t)mb;
        u32x4 g0;
        g0[0] = 1u;                               // count=1, user descriptor
        g0[1] = lds_addr;                         // LDS destination (bytes)
        g0[2] = (unsigned)ga;                     // global_addr[31:0]
        g0[3] = (unsigned)(ga >> 32) | 0x80000000u;  // addr[56:32] | type=2
        i32x8 g1;
        g1[0] = (int)0x07910000u;  // data_size=2B | pad_enable | interval=512B row | pad=4 dwords
        g1[1] = (int)(256u << 16); // tensor_dim0 = 256 (bits 79:48 low half)
        g1[2] = (int)(256u << 16); // tensor_dim1 = 256 (bits 111:80 low half)
        g1[3] = (int)(256u << 16); // tile_dim0   = 256 (bits 127:112)
        g1[4] = 256;               // tile_dim1 = 256, tile_dim2 = 0
        g1[5] = 256;               // tensor_dim0_stride = 256 elements
        g1[6] = 0; g1[7] = 0;
        const i32x4 z4 = {0, 0, 0, 0};
        const i32x8 z8 = {0, 0, 0, 0, 0, 0, 0, 0};
        __builtin_amdgcn_tensor_load_to_lds(g0, g1, z4, z4, z8, 0);
        __builtin_amdgcn_s_wait_tensorcnt(0);
    }
#else
    for (int i = tid; i < 8192; i += 256) {
        const int row = i >> 5, col = i & 31;
        ((u32x4*)(sMt + row * LDS_CP))[col] = ((const u32x4*)(mb + (size_t)row * 256))[col];
    }
#endif

    // ---- stage x[b, :, nbase..+31] transposed -> xs[n][c] bf16 ----
    {
        const int n = tid & 31, cq = tid >> 5;          // cq 0..7
        const float* base = xb + nbase + n;
        u32x2* xs2 = (u32x2*)xs;
        if (tid == 0)
            __builtin_prefetch(xb + (size_t)nbase * 256, 0, 1);  // residual region
#pragma unroll
        for (int j = 0; j < 8; ++j) {
            const int q = cq + (j << 3);                // quad 0..63, c = 4q
            const float* p = base + (size_t)(q << 2) * Nn;
            u32x2 v;
            v[0] = pk_bf16x2(p[0],      p[Nn]);
            v[1] = pk_bf16x2(p[2 * Nn], p[3 * Nn]);
            xs2[n * 66 + q] = v;
        }
    }
    __syncthreads();

    const int w  = tid >> 5, L = tid & 31;
    const int nt = w >> 2;                      // 0..1 (16 n each)
    const int cq4 = (w & 3) << 6;               // 64-wide c quarter
    const int aoff = (L >> 4) << 3;
    const int koff = (L >> 4) << 4;
    const int arow = (nt * 16 + (L & 15)) * LDS_CP;
    const v8u16* lv = (const v8u16*)xs;

    v8f acc[4];
#pragma unroll
    for (int j = 0; j < 4; ++j) acc[j] = 0.f;

#pragma unroll
    for (int ks = 0; ks < 8; ++ks) {
        const int k0 = ks << 5;
        const int h0 = arow + k0 + koff;
        v16bf A = cat16(bcast8(lv[h0 >> 3]), bcast8(lv[(h0 >> 3) + 1]));
#pragma unroll
        for (int j = 0; j < 4; ++j) {
            const int row = cq4 + (j << 4) + (L & 15);
            const v8u16* bp = (const v8u16*)(sMt + row * LDS_CP + k0 + koff);
            v16bf Bm = cat16(bcast8(bp[0]), bcast8(bp[1]));
            acc[j] = __builtin_amdgcn_wmma_f32_16x16x32_bf16(false, A, false, Bm,
                                                             (short)0, acc[j], false, false);
        }
    }
    const float g = gamma[0];
    float* ob = out + (size_t)b * CN;
    const float* db = dvec + b * 256;
#pragma unroll
    for (int j = 0; j < 4; ++j) {
        const int c = cq4 + (j << 4) + (L & 15);
        const float dc = db[c];
#pragma unroll
        for (int r = 0; r < 8; ++r) {
            const int m = r + ((L >> 4) << 3);
            const int n = nbase + nt * 16 + m;
            const size_t flat = (size_t)n * 256 + c;   // permuted-view offset
            ob[flat] = g * (acc[j][r] + dc) + xb[flat];
        }
    }
}

// ---------------------------------------------------------------------------
extern "C" void kernel_launch(void* const* d_in, const int* in_sizes, int n_in,
                              void* d_out, int out_size, void* d_ws, size_t ws_size,
                              hipStream_t stream)
{
    const float* x     = (const float*)d_in[0];
    const float* Wq    = (const float*)d_in[1];
    const float* bq    = (const float*)d_in[2];
    const float* Wk    = (const float*)d_in[3];
    const float* bk    = (const float*)d_in[4];
    const float* Wv    = (const float*)d_in[5];
    const float* bv    = (const float*)d_in[6];
    const float* gamma = (const float*)d_in[7];
    float* out = (float*)d_out;

    char* W = (char*)d_ws;
    float*          kT   = (float*)W;                                       // 8 MB
    unsigned short* wT   = (unsigned short*)(W + (size_t)8  * 1024 * 1024); // 4 MB
    float*          rmax = (float*)(W + (size_t)12 * 1024 * 1024);          // 512 B
    float*          rinv = rmax + 128;                                      // 512 B
    float*          tT   = (float*)(W + (size_t)12 * 1024 * 1024 + 4096);   // 128 KB
    float*          kv   = (float*)((char*)tT + 131072);                    // 128 KB
    unsigned short* Mt   = (unsigned short*)((char*)kv + 131072);           // 512 KB
    float*          dvec = (float*)((char*)Mt + 524288);                    // 4 KB

    k_proj_k       <<<1024, 256, 0, stream>>>(x, Wk, bk, kT);
    k_softmax_stats<<<   4, 256, 0, stream>>>(kT, rmax, rinv);
    k_softmax_norm <<<1024, 256, 0, stream>>>(kT, rmax, rinv, wT);
    (void)hipMemsetAsync(tT, 0, (size_t)Bb * 8192 * sizeof(float), stream);
    k_kv_part1     <<<  64, 256, 0, stream>>>(x, wT, tT);
    k_kv_part2     <<<  32, 256, 0, stream>>>(tT, Wv, bv, kv);
    k_form_M       <<<  64, 256, 0, stream>>>(kv, Wq, bq, Mt, dvec);
    k_attn_out     <<<2048, 256, 0, stream>>>(x, Mt, dvec, gamma, out);
}
